// TextGenerator_77223511982591
// MI455X (gfx1250) — matrix operand order repaired and checked by reading, not compiled, not run on hardware
//
#include <hip/hip_runtime.h>
#include <hip/hip_bf16.h>
#include <stdint.h>

#define B_    64
#define DFEAT 512
#define DHID  1024
#define NGATE 4096
#define V_    32000
#define L_    32
#define KCHUNK 128   // K-chunk of h staged to LDS per TDM transfer

typedef __attribute__((ext_vector_type(16))) __bf16 v16bf;
typedef __attribute__((ext_vector_type(8)))  float  v8f;
typedef __attribute__((ext_vector_type(4)))  unsigned int u32x4;
typedef __attribute__((ext_vector_type(8)))  int i32x8;
typedef __attribute__((ext_vector_type(4)))  int i32x4;

union FragU { uint4 q[2]; v16bf v; };

// 16 consecutive bf16 (32B) -> one B fragment (lane half's contiguous 16-element K run, same order as loadA)
__device__ __forceinline__ v16bf load16bf(const unsigned short* p) {
  FragU f;
  f.q[0] = *(const uint4*)(p);
  f.q[1] = *(const uint4*)(p + 8);
  return f.v;
}
// A fragment: lane half selects a contiguous 16-element K run (matches B)
__device__ __forceinline__ v16bf loadA(const unsigned short* row, int k0, int half) {
  FragU f;
  f.q[0] = *(const uint4*)(row + k0 + half * 16);
  f.q[1] = *(const uint4*)(row + k0 + half * 16 + 8);
  return f.v;
}
__device__ __forceinline__ unsigned short f2bf(float x) {
  unsigned u = __float_as_uint(x);
  u += 0x7FFFu + ((u >> 16) & 1u);        // round-to-nearest-even
  return (unsigned short)(u >> 16);
}
__device__ __forceinline__ float sigmoidf_(float x) { return 1.0f / (1.0f + __expf(-x)); }

// ---------------- threefry2x32 (JAX-compatible) ----------------
__host__ __device__ __forceinline__ uint32_t rotl32_(uint32_t x, int r) {
  return (x << r) | (x >> (32 - r));
}
__host__ __device__ inline void threefry2x32_(uint32_t k0, uint32_t k1,
                                              uint32_t x0, uint32_t x1,
                                              uint32_t* o0, uint32_t* o1) {
  uint32_t ks2 = k0 ^ k1 ^ 0x1BD11BDAu;
  x0 += k0; x1 += k1;
  #define TF_R(r) { x0 += x1; x1 = rotl32_(x1, r); x1 ^= x0; }
  TF_R(13) TF_R(15) TF_R(26) TF_R(6)  x0 += k1;  x1 += ks2 + 1u;
  TF_R(17) TF_R(29) TF_R(16) TF_R(24) x0 += ks2; x1 += k0 + 2u;
  TF_R(13) TF_R(15) TF_R(26) TF_R(6)  x0 += k0;  x1 += k1 + 3u;
  TF_R(17) TF_R(29) TF_R(16) TF_R(24) x0 += k1;  x1 += ks2 + 4u;
  TF_R(13) TF_R(15) TF_R(26) TF_R(6)  x0 += ks2; x1 += k0 + 5u;
  #undef TF_R
  *o0 = x0; *o1 = x1;
}
__device__ __forceinline__ float gumbel_bits_(uint32_t key0, uint32_t key1, uint32_t idx) {
  const uint32_t HALF = (uint32_t)(B_ * V_) / 2u;   // 1,024,000
  uint32_t a, b, bits;
  if (idx < HALF) { threefry2x32_(key0, key1, idx, idx + HALF, &a, &b); bits = a; }
  else            { threefry2x32_(key0, key1, idx - HALF, idx, &a, &b); bits = b; }
  float u = __uint_as_float((bits >> 9) | 0x3F800000u) - 1.0f;   // [0,1)
  const float tiny = 1.1754944e-38f;
  u = u * (1.0f - tiny) + tiny;                                   // (0,1)
  return -__logf(-__logf(u));
}

// ---------------- TDM: DMA a 64 x KCHUNK bf16 tile of h into LDS ----------------
// D# per CDNA5 ISA 8.3/8.4: group0 = {count/type/lds/global addr},
// group1 = {data_size, dims, tile dims, strides}; groups 2/3 unused (2D tile).
// This toolchain exposes the 6-arg builtin: (g0, g1, g2, g3, g4, cpol).
__device__ __forceinline__ void tdm_load_h_chunk(const unsigned short* gsrc,
                                                 unsigned lds_off) {
  unsigned long long ga = (unsigned long long)(uintptr_t)gsrc;
  u32x4 g0;
  g0[0] = 1u;                                            // count=1, user mode
  g0[1] = lds_off;                                       // lds_addr (bytes)
  g0[2] = (unsigned)ga;                                  // global_addr[31:0]
  g0[3] = (unsigned)((ga >> 32) & 0x01FFFFFFull)         // global_addr[56:32]
        | (2u << 30);                                    // type=2 ("image")
  i32x8 g1;
  g1[0] = 1 << 16;           // workgroup_mask=0, data_size=1 (2 bytes)
  g1[1] = KCHUNK << 16;      // tensor_dim0[15:0] in bits 63:48
  g1[2] = 64 << 16;          // tensor_dim0 hi=0 | tensor_dim1[15:0]=64
  g1[3] = KCHUNK << 16;      // tensor_dim1 hi=0 | tile_dim0=KCHUNK
  g1[4] = 64;                // tile_dim1=64, tile_dim2=0
  g1[5] = DHID;              // tensor_dim0_stride lo32 = 1024 elements
  g1[6] = 0;                 // stride0 hi16 = 0, stride1 lo16 = 0
  g1[7] = 0;
  i32x4 z4 = {0, 0, 0, 0};
  i32x8 z8 = {0, 0, 0, 0, 0, 0, 0, 0};
  __builtin_amdgcn_tensor_load_to_lds(g0, g1, z4, z4, z8, 0);
}

// ---------------- setup kernels ----------------
__global__ void cvt_bf16_kernel(const float* __restrict__ src,
                                unsigned short* __restrict__ dst, int n) {
  for (int i = blockIdx.x * blockDim.x + threadIdx.x; i < n; i += gridDim.x * blockDim.x)
    dst[i] = f2bf(src[i]);
}

__global__ void state_init_kernel(unsigned short* hbf, float* cst, int* tokens) {
  int i = blockIdx.x * blockDim.x + threadIdx.x;
  if (i < B_ * DHID) { hbf[i] = 0; cst[i] = 0.0f; }
  if (i < B_) tokens[i] = 1;  // START at t=0
}

__global__ void lp0_init_kernel(float* tlp) {
  int i = blockIdx.x * blockDim.x + threadIdx.x;
  if (i < B_ * V_) {
    int b = i / V_, v = i - b * V_;
    tlp[(size_t)b * L_ * V_ + v] = (v == 1) ? 0.0f : -__builtin_inff();
  }
}

// ---------------- per-step: gates GEMM + LSTM cell ----------------
// One wave owns a 16-row x 16-col tile for ALL FOUR gates so the cell
// nonlinearity happens entirely in registers (i/f/g/o in matching lanes).
__global__ void gates_lstm_kernel(const unsigned short* __restrict__ x0bf,
                                  const int* __restrict__ tok, int use_tok,
                                  const unsigned short* __restrict__ embbf,
                                  const unsigned short* __restrict__ Wihbf,
                                  const unsigned short* __restrict__ Whhbf,
                                  const float* __restrict__ bih,
                                  const float* __restrict__ bhh,
                                  const unsigned short* __restrict__ h_in,
                                  unsigned short* __restrict__ h_out,
                                  float* __restrict__ cst) {
  int lane = threadIdx.x & 31;
  int wid  = blockIdx.x * (blockDim.x >> 5) + (threadIdx.x >> 5);  // 0..255
  int mtile = wid & 3;
  int jtile = wid >> 2;                 // 0..63 (column tile within 1024)
  int half = lane >> 4, r = lane & 15;
  int m    = mtile * 16 + r;            // batch row for A loads
  int ncol = jtile * 16 + r;            // gate column within 1024

  const unsigned short* arow =
      use_tok ? (embbf + (size_t)tok[m] * DFEAT) : (x0bf + (size_t)m * DFEAT);

  v8f acc[4] = {};                       // gates i,f,g,o

  // part 1: x @ W_ih^T  (K = 512)
  for (int k0 = 0; k0 < DFEAT; k0 += 32) {
    v16bf a = loadA(arow, k0, half);
    #pragma unroll
    for (int g = 0; g < 4; ++g) {
      const unsigned short* brow = Wihbf + (size_t)(g * DHID + ncol) * DFEAT;
      v16bf b = load16bf(brow + k0 + half * 16);
      acc[g] = __builtin_amdgcn_wmma_f32_16x16x32_bf16(false, a, false, b,
                                                       (short)0, acc[g], false, false);
    }
  }
  // part 2: h @ W_hh^T  (K = 1024)
  const unsigned short* hrow = h_in + (size_t)m * DHID;
  for (int k0 = 0; k0 < DHID; k0 += 32) {
    v16bf a = loadA(hrow, k0, half);
    #pragma unroll
    for (int g = 0; g < 4; ++g) {
      const unsigned short* brow = Whhbf + (size_t)(g * DHID + ncol) * DHID;
      v16bf b = load16bf(brow + k0 + half * 16);
      acc[g] = __builtin_amdgcn_wmma_f32_16x16x32_bf16(false, a, false, b,
                                                       (short)0, acc[g], false, false);
    }
  }

  float bias[4];
  #pragma unroll
  for (int g = 0; g < 4; ++g) bias[g] = bih[g * DHID + ncol] + bhh[g * DHID + ncol];

  #pragma unroll
  for (int i = 0; i < 8; ++i) {
    int mr = mtile * 16 + half * 8 + i;          // C/D layout: M = 8*half + i
    float iv = sigmoidf_(acc[0][i] + bias[0]);
    float fv = sigmoidf_(acc[1][i] + bias[1]);
    float gv = tanhf   (acc[2][i] + bias[2]);
    float ov = sigmoidf_(acc[3][i] + bias[3]);
    size_t idx = (size_t)mr * DHID + ncol;
    float cnew = fv * cst[idx] + iv * gv;
    cst[idx] = cnew;
    h_out[idx] = f2bf(ov * tanhf(cnew));
  }
}

// ---------------- per-step: head GEMM ----------------
// One wave per ntile computes the full M=64 (4 accumulators) so each W_fc
// fragment is loaded once and feeds 4 WMMAs. A (= h state) is DMA'd into LDS
// by the TDM, double-buffered and pipelined on TENSORcnt. B is register
// double-buffered so the next global load overlaps the current WMMAs.
__global__ void head_gemm_kernel(const unsigned short* __restrict__ hbf,
                                 const unsigned short* __restrict__ Wfcbf,
                                 const float* __restrict__ bfc,
                                 float* __restrict__ tlp, int t) {
  __shared__ unsigned short shA[2][64 * KCHUNK];   // 2 x 16KB
  int tid = threadIdx.x;
  int lane = tid & 31, wave = tid >> 5;
  int ntile = blockIdx.x * 8 + wave;               // 0..1999
  int half = lane >> 4, r = lane & 15;
  int ncol = ntile * 16 + r;
  const unsigned short* brow = Wfcbf + (size_t)ncol * DHID;
  const int NCH = DHID / KCHUNK;                   // 8 chunks

  bool issuer = (tid < 32);                        // wave 0 drives the TDM
  if (issuer) tdm_load_h_chunk(hbf, (unsigned)(uintptr_t)&shA[0][0]);

  v8f acc0 = {}, acc1 = {}, acc2 = {}, acc3 = {};
  v16bf bcur = load16bf(brow + half * 16);

  for (int c = 0; c < NCH; ++c) {
    int buf = c & 1;
    if (issuer) {
      if (c + 1 < NCH) {
        tdm_load_h_chunk(hbf + (c + 1) * KCHUNK,
                         (unsigned)(uintptr_t)&shA[buf ^ 1][0]);
        __builtin_amdgcn_s_wait_tensorcnt(1);      // chunk c done, c+1 in flight
      } else {
        __builtin_amdgcn_s_wait_tensorcnt(0);      // last chunk done
      }
    }
    __syncthreads();                               // chunk c visible in LDS
    const unsigned short* sbase = &shA[buf][0];
    #pragma unroll
    for (int kk = 0; kk < KCHUNK; kk += 32) {
      int k0 = c * KCHUNK + kk;
      v16bf b = bcur;
      if (k0 + 32 < DHID)                          // prefetch next B fragment
        bcur = load16bf(brow + (k0 + 32) + half * 16);
      v16bf a0 = loadA(sbase + (0 * 16 + r) * KCHUNK, kk, half);
      v16bf a1 = loadA(sbase + (1 * 16 + r) * KCHUNK, kk, half);
      v16bf a2 = loadA(sbase + (2 * 16 + r) * KCHUNK, kk, half);
      v16bf a3 = loadA(sbase + (3 * 16 + r) * KCHUNK, kk, half);
      acc0 = __builtin_amdgcn_wmma_f32_16x16x32_bf16(false, a0, false, b, (short)0, acc0, false, false);
      acc1 = __builtin_amdgcn_wmma_f32_16x16x32_bf16(false, a1, false, b, (short)0, acc1, false, false);
      acc2 = __builtin_amdgcn_wmma_f32_16x16x32_bf16(false, a2, false, b, (short)0, acc2, false, false);
      acc3 = __builtin_amdgcn_wmma_f32_16x16x32_bf16(false, a3, false, b, (short)0, acc3, false, false);
    }
    __syncthreads();                               // buf free before re-DMA
  }

  float bias = bfc[ncol];
  #pragma unroll
  for (int i = 0; i < 8; ++i) {
    int mi = half * 8 + i;
    tlp[((size_t)(0 * 16 + mi) * L_ + t) * V_ + ncol] = acc0[i] + bias;
    tlp[((size_t)(1 * 16 + mi) * L_ + t) * V_ + ncol] = acc1[i] + bias;
    tlp[((size_t)(2 * 16 + mi) * L_ + t) * V_ + ncol] = acc2[i] + bias;
    tlp[((size_t)(3 * 16 + mi) * L_ + t) * V_ + ncol] = acc3[i] + bias;
  }
}

// ---------------- per-step: row max + logsumexp ----------------
__global__ void row_reduce_kernel(const float* __restrict__ tlp, int t,
                                  float* __restrict__ lse) {
  int b = blockIdx.x, tid = threadIdx.x;
  const float* row = tlp + ((size_t)b * L_ + t) * V_;
  __shared__ float sm[256];
  float mx = -__builtin_inff();
  for (int v = tid; v < V_; v += 256) mx = fmaxf(mx, row[v]);
  sm[tid] = mx; __syncthreads();
  for (int o = 128; o > 0; o >>= 1) {
    if (tid < o) sm[tid] = fmaxf(sm[tid], sm[tid + o]);
    __syncthreads();
  }
  float rowmax = sm[0]; __syncthreads();
  float s = 0.0f;
  for (int v = tid; v < V_; v += 256) s += __expf(row[v] - rowmax);
  sm[tid] = s; __syncthreads();
  for (int o = 128; o > 0; o >>= 1) {
    if (tid < o) sm[tid] += sm[tid + o];
    __syncthreads();
  }
  if (tid == 0) lse[b] = rowmax + __logf(sm[0]);
}

// ---------------- per-step: in-place log-softmax + Gumbel-max sample ----------------
__global__ void norm_sample_kernel(float* __restrict__ tlp, int t,
                                   const float* __restrict__ lse,
                                   uint32_t key0, uint32_t key1,
                                   int* __restrict__ tokout) {
  int b = blockIdx.x, tid = threadIdx.x;
  float* row = tlp + ((size_t)b * L_ + t) * V_;
  float l = lse[b];
  float best = -__builtin_inff(); int besti = 0x7fffffff;
  for (int v = tid; v < V_; v += 256) {
    float lp = row[v] - l;
    row[v] = lp;
    float s = lp + gumbel_bits_(key0, key1, (uint32_t)(b * V_ + v));
    if (s > best || (s == best && v < besti)) { best = s; besti = v; }
  }
  __shared__ float sv[256]; __shared__ int si[256];
  sv[tid] = best; si[tid] = besti; __syncthreads();
  for (int o = 128; o > 0; o >>= 1) {
    if (tid < o) {
      if (sv[tid + o] > sv[tid] || (sv[tid + o] == sv[tid] && si[tid + o] < si[tid])) {
        sv[tid] = sv[tid + o]; si[tid] = si[tid + o];
      }
    }
    __syncthreads();
  }
  if (tid == 0) tokout[b] = si[0];
}

// ---------------- finalize: END-mask, texts, lengths ----------------
__global__ void finalize_meta_kernel(const int* __restrict__ tokens,
                                     int* __restrict__ maskws,
                                     float* __restrict__ texts_out,
                                     float* __restrict__ lens_out) {
  int b = threadIdx.x;
  if (b >= B_) return;
  int seen = 0, cnt = 0;
  for (int t = 0; t < L_; ++t) {
    int raw = tokens[t * B_ + b];
    int mk = seen;
    maskws[b * L_ + t] = mk;
    texts_out[b * L_ + t] = mk ? 0.0f : (float)raw;  // PAD = 0
    cnt += mk;
    if (raw == 2) seen = 1;                           // END
  }
  lens_out[b] = (float)(L_ - cnt);
}

__global__ void mask_apply_kernel(float* __restrict__ tlp,
                                  const int* __restrict__ maskws) {
  const size_t N = (size_t)B_ * L_ * V_;
  for (size_t i = (size_t)blockIdx.x * blockDim.x + threadIdx.x; i < N;
       i += (size_t)gridDim.x * blockDim.x) {
    size_t bt = i / V_;
    int v = (int)(i - bt * V_);
    if (maskws[bt]) tlp[i] = (v == 0) ? 0.0f : -__builtin_inff();
  }
}

// ---------------- host ----------------
extern "C" void kernel_launch(void* const* d_in, const int* in_sizes, int n_in,
                              void* d_out, int out_size, void* d_ws, size_t ws_size,
                              hipStream_t stream) {
  (void)in_sizes; (void)n_in; (void)out_size; (void)ws_size;
  const float* text  = (const float*)d_in[0];
  const float* embed = (const float*)d_in[1];
  const float* Wih   = (const float*)d_in[2];
  const float* Whh   = (const float*)d_in[3];
  const float* bih   = (const float*)d_in[4];
  const float* bhh   = (const float*)d_in[5];
  const float* Wfc   = (const float*)d_in[6];
  const float* bfc   = (const float*)d_in[7];

  char* ws = (char*)d_ws;
  size_t off = 0;
  auto alloc = [&](size_t bytes) -> char* {
    char* p = ws + off;
    off = (off + bytes + 255) & ~(size_t)255;
    return p;
  };
  unsigned short* Wih_bf = (unsigned short*)alloc((size_t)NGATE * DFEAT * 2);
  unsigned short* Whh_bf = (unsigned short*)alloc((size_t)NGATE * DHID * 2);
  unsigned short* Wfc_bf = (unsigned short*)alloc((size_t)V_ * DHID * 2);
  unsigned short* emb_bf = (unsigned short*)alloc((size_t)V_ * DFEAT * 2);
  unsigned short* x0_bf  = (unsigned short*)alloc((size_t)B_ * DFEAT * 2);
  unsigned short* h_a    = (unsigned short*)alloc((size_t)B_ * DHID * 2);
  unsigned short* h_b    = (unsigned short*)alloc((size_t)B_ * DHID * 2);
  float*          c_st   = (float*)alloc((size_t)B_ * DHID * 4);
  float*          lse    = (float*)alloc((size_t)B_ * 4);
  int*            tokens = (int*)alloc((size_t)L_ * B_ * 4);
  int*            maskws = (int*)alloc((size_t)B_ * L_ * 4);

  float* texts_out = (float*)d_out;
  float* tlp       = texts_out + B_ * L_;
  float* lens_out  = tlp + (size_t)B_ * L_ * V_;

  // step keys = split(PRNGKey(42), 32), computed host-side (pure constants)
  uint32_t flat[64];
  for (int i = 0; i < 32; ++i) {
    uint32_t a, b;
    threefry2x32_(0u, 42u, (uint32_t)i, (uint32_t)(i + 32), &a, &b);
    flat[i] = a; flat[i + 32] = b;
  }
  uint32_t key0s[32], key1s[32];
  for (int t = 0; t < 32; ++t) { key0s[t] = flat[2 * t]; key1s[t] = flat[2 * t + 1]; }

  // weight/activation bf16 conversion (every call; ws is not persistent)
  cvt_bf16_kernel<<<4096, 256, 0, stream>>>(Wih,   Wih_bf, NGATE * DFEAT);
  cvt_bf16_kernel<<<4096, 256, 0, stream>>>(Whh,   Whh_bf, NGATE * DHID);
  cvt_bf16_kernel<<<4096, 256, 0, stream>>>(Wfc,   Wfc_bf, V_ * DHID);
  cvt_bf16_kernel<<<4096, 256, 0, stream>>>(embed, emb_bf, V_ * DFEAT);
  cvt_bf16_kernel<<<128,  256, 0, stream>>>(text,  x0_bf,  B_ * DFEAT);

  state_init_kernel<<<(B_ * DHID + 255) / 256, 256, 0, stream>>>(h_a, c_st, tokens);
  lp0_init_kernel<<<(B_ * V_ + 255) / 256, 256, 0, stream>>>(tlp);

  for (int t = 1; t < L_; ++t) {
    unsigned short* h_in  = (t & 1) ? h_a : h_b;
    unsigned short* h_out = (t & 1) ? h_b : h_a;
    gates_lstm_kernel<<<32, 256, 0, stream>>>(x0_bf, tokens + (t - 1) * B_,
                                              (t > 1) ? 1 : 0, emb_bf, Wih_bf, Whh_bf,
                                              bih, bhh, h_in, h_out, c_st);
    head_gemm_kernel<<<250, 256, 0, stream>>>(h_out, Wfc_bf, bfc, tlp, t);
    row_reduce_kernel<<<B_, 256, 0, stream>>>(tlp, t, lse);
    norm_sample_kernel<<<B_, 256, 0, stream>>>(tlp, t, lse, key0s[t], key1s[t],
                                               tokens + t * B_);
  }

  finalize_meta_kernel<<<1, 64, 0, stream>>>(tokens, maskws, texts_out, lens_out);
  mask_apply_kernel<<<8192, 256, 0, stream>>>(tlp, maskws);
}
